// GDEFunc_47536698032170
// MI455X (gfx1250) — matrix operand order repaired and checked by reading, not compiled, not run on hardware
//
#include <hip/hip_runtime.h>
#include <hip/hip_bf16.h>
#include <stdint.h>

typedef _Float16 v16h __attribute__((ext_vector_type(16)));
typedef _Float16 v8h  __attribute__((ext_vector_type(8)));
typedef float    v8f  __attribute__((ext_vector_type(8)));

// ---------------- utility kernels ----------------

__global__ void zero_f32(float* __restrict__ p, long n) {
  long i = blockIdx.x * (long)blockDim.x + threadIdx.x;
  long stride = (long)gridDim.x * blockDim.x;
  for (; i < n; i += stride) p[i] = 0.0f;
}

// One thread per (edge, 4-float group): gather z[src], atomic-add into agg[dst].
__global__ void scatter_edges(const float* __restrict__ z,
                              const long long* __restrict__ ei,
                              float* __restrict__ agg,
                              float* __restrict__ deg,
                              int E, int D) {
  const int groups = D >> 2;                      // 32 groups of float4
  long t = blockIdx.x * (long)blockDim.x + threadIdx.x;
  long total = (long)E * groups;
  if (t >= total) return;
  int e = (int)(t / groups);
  int g = (int)(t % groups);
  long long s = ei[e];
  long long d = ei[(long)E + e];
  const float4 v = *(const float4*)(z + (long)s * D + (g << 2));
  float* a = agg + (long)d * D + (g << 2);
  __hip_atomic_fetch_add(a + 0, v.x, __ATOMIC_RELAXED, __HIP_MEMORY_SCOPE_AGENT);
  __hip_atomic_fetch_add(a + 1, v.y, __ATOMIC_RELAXED, __HIP_MEMORY_SCOPE_AGENT);
  __hip_atomic_fetch_add(a + 2, v.z, __ATOMIC_RELAXED, __HIP_MEMORY_SCOPE_AGENT);
  __hip_atomic_fetch_add(a + 3, v.w, __ATOMIC_RELAXED, __HIP_MEMORY_SCOPE_AGENT);
  if (g == 0)
    __hip_atomic_fetch_add(deg + d, 1.0f, __ATOMIC_RELAXED, __HIP_MEMORY_SCOPE_AGENT);
}

// Acat[i][0:128]   = f16(agg[i]/max(deg[i],1))
// Acat[i][128:256] = f16(z[i])
__global__ void prep_acat(const float* __restrict__ agg,
                          const float* __restrict__ deg,
                          const float* __restrict__ z,
                          _Float16* __restrict__ Acat,
                          int N, int D) {
  long t = blockIdx.x * (long)blockDim.x + threadIdx.x;
  long total = (long)N * D;
  if (t >= total) return;
  int i = (int)(t / D);
  int d = (int)(t % D);
  float dg = fmaxf(deg[i], 1.0f);
  long base = (long)i * (2 * D);
  Acat[base + d]     = (_Float16)(agg[t] / dg);
  Acat[base + D + d] = (_Float16)z[t];
}

// dst[n*ld + off + k] = f16(src[k*Nn + n])  : transpose+convert weights
__global__ void conv_transpose(_Float16* __restrict__ dst,
                               const float* __restrict__ src,
                               int K, int Nn, int ld, int off) {
  long t = blockIdx.x * (long)blockDim.x + threadIdx.x;
  long total = (long)K * Nn;
  if (t >= total) return;
  int k = (int)(t / Nn);
  int n = (int)(t % Nn);
  dst[(long)n * ld + off + k] = (_Float16)src[(long)k * Nn + n];
}

// ---------------- WMMA GEMM: Out = act(A @ W + bias) ----------------
// A:  f16 row-major [Nrows][Kin]
// Wt: f16 [Nout][Kin]  (transposed weight -> contiguous B columns)
// Each wave computes one 16-row tile x NT 16-col tiles: A frag loaded once
// per k-step, reused for NT WMMAs; NT accumulators hide WMMA->WMMA hazards.
// ACT: 0 = relu, 1 = tanh, 2 = identity
template <int ACT, bool OUTF32, int NT>
__global__ void gemm_wmma_f16(const _Float16* __restrict__ A, int Kin,
                              const _Float16* __restrict__ Wt,
                              const float* __restrict__ bias,
                              void* __restrict__ Out,
                              int Nout, int Nrows) {
  const int wave = (int)((blockIdx.x * (long)blockDim.x + threadIdx.x) >> 5);
  const int lane = threadIdx.x & 31;
  const int ngrp = Nout / (16 * NT);          // column-tile groups
  const int tm   = wave / ngrp;
  const int tg   = wave - tm * ngrp;
  const int row0 = tm << 4;
  if (row0 >= Nrows) return;                  // wave-uniform: EXEC stays all-1s
  const int col0  = tg * (16 * NT);
  const int half_ = lane >> 4;                // 0 or 1
  const int l16   = lane & 15;

  // Out-of-range A rows only affect C rows that are never stored, so just
  // clamp the pointer into bounds (no masking needed).
  int arow = row0 + l16;
  if (arow >= Nrows) arow = Nrows - 1;
  const _Float16* Ap = A + (long)arow * Kin;

  const _Float16* Bp[NT];
#pragma unroll
  for (int j = 0; j < NT; ++j)
    Bp[j] = Wt + (long)(col0 + j * 16 + l16) * Kin;

  v8f c[NT];
#pragma unroll
  for (int j = 0; j < NT; ++j) c[j] = (v8f){};

  for (int k0 = 0; k0 < Kin; k0 += 32) {
    // A frag (ISA layout): elem e<8 -> K = 8*half + e ; e>=8 -> K = 16 + 8*half + (e-8)
    v8h alo = *(const v8h*)(Ap + k0 + 8 * half_);
    v8h ahi = *(const v8h*)(Ap + k0 + 16 + 8 * half_);
    v16h a;
#pragma unroll
    for (int i = 0; i < 8; ++i) { a[i] = alo[i]; a[8 + i] = ahi[i]; }

#pragma unroll
    for (int j = 0; j < NT; ++j) {
      // B frag: elem e<8 -> K = 8*half + e ; e>=8 -> K = 16 + 8*half + (e-8), column = lane%16 (contiguous in Wt row)
      v8h blo = *(const v8h*)(Bp[j] + k0 + 8 * half_);
      v8h bhi = *(const v8h*)(Bp[j] + k0 + 16 + 8 * half_);
      v16h b;
#pragma unroll
      for (int i = 0; i < 8; ++i) { b[i] = blo[i]; b[8 + i] = bhi[i]; }
      c[j] = __builtin_amdgcn_wmma_f32_16x16x32_f16(
          /*neg_a=*/false, a, /*neg_b=*/false, b,
          /*c_mod=*/(short)0, c[j], /*reuse_a=*/false, /*reuse_b=*/false);
    }
  }

#pragma unroll
  for (int j = 0; j < NT; ++j) {
    const int cj = col0 + j * 16 + l16;
    const float bv = bias[cj];
#pragma unroll
    for (int i = 0; i < 8; ++i) {
      int r = row0 + i + 8 * half_;           // C layout: VGPR i -> M = i + 8*half
      if (r >= Nrows) continue;
      float v = c[j][i] + bv;
      if (ACT == 0)      v = fmaxf(v, 0.0f);
      else if (ACT == 1) v = tanhf(v);
      long idx = (long)r * Nout + cj;
      if (OUTF32) ((float*)Out)[idx]    = v;
      else        ((_Float16*)Out)[idx] = (_Float16)v;
    }
  }
}

// ---------------- host side ----------------

static inline int gemm_blocks(int nrows, int nout, int nt) {
  long waves = ((long)(nrows + 15) / 16) * (nout / (16 * nt));
  return (int)((waves * 32 + 255) / 256);
}

extern "C" void kernel_launch(void* const* d_in, const int* in_sizes, int n_in,
                              void* d_out, int out_size, void* d_ws, size_t ws_size,
                              hipStream_t stream) {
  const float*     z   = (const float*)d_in[0];
  const long long* ei  = (const long long*)d_in[1];
  const float*     Wg  = (const float*)d_in[2];
  const float*     Ws  = (const float*)d_in[3];
  const float*     bg  = (const float*)d_in[4];
  const float*     W1  = (const float*)d_in[5];
  const float*     b1  = (const float*)d_in[6];
  const float*     W2  = (const float*)d_in[7];
  const float*     b2  = (const float*)d_in[8];
  const float*     W3  = (const float*)d_in[9];
  const float*     b3  = (const float*)d_in[10];

  const int D = 128;
  const int N = in_sizes[0] / D;      // 100000
  const int E = in_sizes[1] / 2;      // 600000
  const int H = in_sizes[6];          // 512 (b1 length)

  // ---- workspace layout (256B aligned) ----
  char* base = (char*)d_ws;
  size_t o = 0;
  auto take = [&](size_t bytes) { size_t r = o; o = (o + bytes + 255) & ~(size_t)255; return r; };
  size_t o_agg  = take((size_t)N * D * sizeof(float));        // 51.2 MB
  size_t o_acat = take((size_t)N * 2 * D * sizeof(_Float16)); // 51.2 MB
  size_t o_h0   = take((size_t)N * D * sizeof(_Float16));     // 25.6 MB
  size_t o_h1   = take((size_t)N * H * sizeof(_Float16));     // 102.4 MB
  size_t o_deg  = take((size_t)N * sizeof(float));
  size_t o_wc   = take((size_t)2 * D * D * sizeof(_Float16)); // WcatT [128][256]
  size_t o_w1   = take((size_t)H * D * sizeof(_Float16));     // W1T   [512][128]
  size_t o_w2   = take((size_t)H * H * sizeof(_Float16));     // W2T   [512][512]
  size_t o_w3   = take((size_t)D * H * sizeof(_Float16));     // W3T   [128][512]
  size_t o_h2   = o_agg;  // h2 (N*H f16 = 102.4MB) reuses dead agg+Acat region

  float*    agg  = (float*)(base + o_agg);
  float*    deg  = (float*)(base + o_deg);
  _Float16* Acat = (_Float16*)(base + o_acat);
  _Float16* h0   = (_Float16*)(base + o_h0);
  _Float16* h1   = (_Float16*)(base + o_h1);
  _Float16* h2   = (_Float16*)(base + o_h2);
  _Float16* WcT  = (_Float16*)(base + o_wc);
  _Float16* W1T  = (_Float16*)(base + o_w1);
  _Float16* W2T  = (_Float16*)(base + o_w2);
  _Float16* W3T  = (_Float16*)(base + o_w3);

  // 1) zero accumulators
  zero_f32<<<2048, 256, 0, stream>>>(agg, (long)N * D);
  zero_f32<<<512,  256, 0, stream>>>(deg, (long)N);

  // 2) edge scatter (hardware FP32 atomics)
  {
    long total = (long)E * (D / 4);
    int blocks = (int)((total + 255) / 256);
    scatter_edges<<<blocks, 256, 0, stream>>>(z, ei, agg, deg, E, D);
  }

  // 3) weight transpose+convert to f16
  {
    int t = D * D;   // 16384
    conv_transpose<<<(t + 255) / 256, 256, 0, stream>>>(WcT, Wg, D, D, 2 * D, 0);
    conv_transpose<<<(t + 255) / 256, 256, 0, stream>>>(WcT, Ws, D, D, 2 * D, D);
    t = D * H;       // 65536
    conv_transpose<<<(t + 255) / 256, 256, 0, stream>>>(W1T, W1, D, H, D, 0);
    t = H * H;       // 262144
    conv_transpose<<<(t + 255) / 256, 256, 0, stream>>>(W2T, W2, H, H, H, 0);
    t = H * D;       // 65536
    conv_transpose<<<(t + 255) / 256, 256, 0, stream>>>(W3T, W3, H, D, H, 0);
  }

  // 4) mean-aggregate + concat -> f16 activations
  {
    long total = (long)N * D;
    prep_acat<<<(int)((total + 255) / 256), 256, 0, stream>>>(agg, deg, z, Acat, N, D);
  }

  // 5) GEMM chain (all WMMA f16 -> f32 accumulate), NT=8 column tiles/wave:
  //    A fragment is loaded once per k-step and reused for 8 WMMAs; for the
  //    128-wide outputs this means each A row is read exactly once.
  // h0 = relu([aggm, z] @ [Wg; Ws] + bg)   (N x 256) @ (256 x 128)
  gemm_wmma_f16<0, false, 8><<<gemm_blocks(N, D, 8), 256, 0, stream>>>(
      Acat, 2 * D, WcT, bg, h0, D, N);
  // h1 = tanh(h0 @ W1 + b1)                (N x 128) @ (128 x 512)
  gemm_wmma_f16<1, false, 8><<<gemm_blocks(N, H, 8), 256, 0, stream>>>(
      h0, D, W1T, b1, h1, H, N);
  // h2 = tanh(h1 @ W2 + b2)                (N x 512) @ (512 x 512)
  gemm_wmma_f16<1, false, 8><<<gemm_blocks(N, H, 8), 256, 0, stream>>>(
      h1, H, W2T, b2, h2, H, N);
  // out = h2 @ W3 + b3                     (N x 512) @ (512 x 128) -> f32
  gemm_wmma_f16<2, true, 8><<<gemm_blocks(N, D, 8), 256, 0, stream>>>(
      h2, H, W3T, b3, d_out, D, N);

  (void)n_in; (void)out_size; (void)ws_size;
}
